// AttentionTest_14053132993039
// MI455X (gfx1250) — matrix operand order repaired and checked by reading, not compiled, not run on hardware
//
#include <hip/hip_runtime.h>

// Problem constants from the reference
#define B_ 64
#define T_ 4096
#define D_ 128
#define S_ 256

typedef _Float16 v16h __attribute__((ext_vector_type(16)));
typedef _Float16 v8h  __attribute__((ext_vector_type(8)));
typedef _Float16 v4h  __attribute__((ext_vector_type(4)));
typedef _Float16 v2h  __attribute__((ext_vector_type(2)));
typedef __fp16   v2hf __attribute__((ext_vector_type(2)));
typedef float    v8f  __attribute__((ext_vector_type(8)));

// 128 + 8 halves padding: row stride 272 B shifts consecutive rows by 4 LDS
// banks, so 16 lanes reading 16 consecutive rows at the same d-offset are
// conflict-free (plain 256 B stride would be a 16-way conflict).
#define WO_STRIDE 136

// v_cvt_pk_rtz_f16_f32: two f32 -> packed 2xf16 in one VALU op.
// The builtin returns an __fp16 vector; bit-cast to our _Float16 vector type.
__device__ __forceinline__ v2h cvt_pk(float x, float y) {
  v2hf t = __builtin_amdgcn_cvt_pkrtz(x, y);
  v2h r;
  __builtin_memcpy(&r, &t, sizeof(r));
  return r;
}

// Branch-free tanh on hardware transcendentals:
//   tanh(x) = 1 - 2/(e^(2x)+1);  e^(2x) = exp2(x * 2*log2(e))
// |x| large positive: exp2 -> +inf -> rcp -> 0 -> +1.
// |x| large negative: exp2 -> 0   -> 1-2/1  -> -1.  No branches, no EXEC ops.
__device__ __forceinline__ float fast_tanh(float x) {
  float e2x = __builtin_amdgcn_exp2f(x * 2.8853900817779268f);
  return 1.0f - 2.0f * __builtin_amdgcn_rcpf(e2x + 1.0f);
}

__global__ __launch_bounds__(256)
void attn_score_wmma_kernel(const float* __restrict__ X,   // [B,T,D]
                            const float* __restrict__ g,   // [B,D]
                            const float* __restrict__ WO,  // [S,D]
                            const float* __restrict__ WG,  // [S,D]
                            const float* __restrict__ v,   // [S]
                            float* __restrict__ out) {     // [B,T]
  __shared__ _Float16 lds_wo[S_ * WO_STRIDE]; // WO as f16, row-major padded
  __shared__ float2   lds_tv[S_];             // {term2[s], v[s]}

  const int tid = threadIdx.x;
  const int b   = blockIdx.y;

  // ---- Prologue 1: WO fp32 -> f16 into LDS (WO is L2-resident, 128 KB) ----
  const float4* wo4 = (const float4*)WO;
  for (int i = tid; i < S_ * D_ / 4; i += 256) {
    int s = i >> 5;        // 32 float4 per row of 128
    int j = i & 31;
    float4 w = wo4[i];
    v2h p0 = cvt_pk(w.x, w.y);
    v2h p1 = cvt_pk(w.z, w.w);
    v4h h  = __builtin_shufflevector(p0, p1, 0, 1, 2, 3);
    *(v4h*)&lds_wo[s * WO_STRIDE + j * 4] = h;
  }

  // ---- Prologue 2: term2[s] = WG[s,:] . g[b,:]; stage {term2, v} ----
  {
    int s = tid;  // 256 threads == S
    const float4* gg = (const float4*)(g + (size_t)b * D_);
    const float4* wg = (const float4*)(WG + (size_t)s * D_);
    float acc = 0.f;
#pragma unroll
    for (int k = 0; k < D_ / 4; ++k) {
      float4 a = wg[k], c = gg[k];
      acc += a.x * c.x + a.y * c.y + a.z * c.z + a.w * c.w;
    }
    lds_tv[s] = make_float2(acc, v[s]);
  }
  __syncthreads();

  const int wave = tid >> 5;           // 8 waves per block
  const int lane = tid & 31;
  const int lo   = lane & 15;
  const int hi   = lane >> 4;          // lane half
  const int t0   = blockIdx.x * 128 + wave * 16;
  const int row  = t0 + lo;

  // ---- A fragments: X[row,:] -> f16, CDNA5 16-bit A layout (16x32 per kc):
  // lanes 0-15 hold K = {kc*32+0..7, kc*32+16..23}, lanes 16-31 the +8 halves.
  v16h afrag[4];
  const float* xrow = X + ((size_t)b * T_ + row) * D_;
#pragma unroll
  for (int kc = 0; kc < 4; ++kc) {
    const float4* ap = (const float4*)(xrow + kc * 32 + hi * 8);
    float4 a0 = ap[0], a1 = ap[1];   // K = hi*8 + 0..7
    float4 a2 = ap[4], a3 = ap[5];   // K = 16 + hi*8 + 0..7
    v2h p0 = cvt_pk(a0.x, a0.y);
    v2h p1 = cvt_pk(a0.z, a0.w);
    v2h p2 = cvt_pk(a1.x, a1.y);
    v2h p3 = cvt_pk(a1.z, a1.w);
    v2h p4 = cvt_pk(a2.x, a2.y);
    v2h p5 = cvt_pk(a2.z, a2.w);
    v2h p6 = cvt_pk(a3.x, a3.y);
    v2h p7 = cvt_pk(a3.z, a3.w);
    v4h q0 = __builtin_shufflevector(p0, p1, 0, 1, 2, 3);
    v4h q1 = __builtin_shufflevector(p2, p3, 0, 1, 2, 3);
    v4h q2 = __builtin_shufflevector(p4, p5, 0, 1, 2, 3);
    v4h q3 = __builtin_shufflevector(p6, p7, 0, 1, 2, 3);
    v8h r0 = __builtin_shufflevector(q0, q1, 0, 1, 2, 3, 4, 5, 6, 7);
    v8h r1 = __builtin_shufflevector(q2, q3, 0, 1, 2, 3, 4, 5, 6, 7);
    afrag[kc] = __builtin_shufflevector(r0, r1,
        0, 1, 2, 3, 4, 5, 6, 7, 8, 9, 10, 11, 12, 13, 14, 15);
  }

  float sacc[8];
#pragma unroll
  for (int r = 0; r < 8; ++r) sacc[r] = 0.f;

  // ---- Sweep the 16 N-tiles of S; 4 chained WMMAs (K=128) per tile ----
#pragma unroll 4
  for (int nt = 0; nt < 16; ++nt) {
    const int scol = nt * 16 + lo;   // B fragment column = WO row
    v8f acc = {};
#pragma unroll
    for (int kc = 0; kc < 4; ++kc) {
      // B layout (32x16 per kc): lanes 0-15 hold K={0..7,16..23}, lanes 16-31 the +8 halves
      const _Float16* bp = &lds_wo[scol * WO_STRIDE + kc * 32 + hi * 8];
      v8h blo = *(const v8h*)bp;
      v8h bhi = *(const v8h*)(bp + 16);
      v16h bfrag = __builtin_shufflevector(blo, bhi,
          0, 1, 2, 3, 4, 5, 6, 7, 8, 9, 10, 11, 12, 13, 14, 15);
      acc = __builtin_amdgcn_wmma_f32_16x16x32_f16(
          /*neg_a=*/false, afrag[kc], /*neg_b=*/false, bfrag,
          /*c_mod=*/(short)0, acc, /*reuse_a=*/false, /*reuse_b=*/false);
    }
    // Fused epilogue: v[s] * tanh(term1 + term2), accumulated per lane.
    float2 tv = lds_tv[scol];
#pragma unroll
    for (int r = 0; r < 8; ++r)
      sacc[r] += tv.y * fast_tanh(acc[r] + tv.x);
  }

  // ---- Reduce across the 16 lanes of each half (C layout: VGPR r holds
  // row M=r for lanes 0-15, M=r+8 for lanes 16-31; N = lane&15) ----
#pragma unroll
  for (int r = 0; r < 8; ++r) {
#pragma unroll
    for (int m = 1; m < 16; m <<= 1)
      sacc[r] += __shfl_xor(sacc[r], m, 16);
  }
  if (lo == 0) {
    float* orow = out + (size_t)b * T_ + t0 + hi * 8;
#pragma unroll
    for (int r = 0; r < 8; ++r) orow[r] = sacc[r];
  }
}

extern "C" void kernel_launch(void* const* d_in, const int* in_sizes, int n_in,
                              void* d_out, int out_size, void* d_ws, size_t ws_size,
                              hipStream_t stream) {
  (void)in_sizes; (void)n_in; (void)out_size; (void)d_ws; (void)ws_size;
  const float* X  = (const float*)d_in[0];  // [B,T,D]
  const float* g  = (const float*)d_in[1];  // [B,D]
  const float* WO = (const float*)d_in[2];  // [S,D]
  const float* WG = (const float*)d_in[3];  // [S,D]
  const float* v  = (const float*)d_in[4];  // [1,S]
  float* out = (float*)d_out;               // [B,T]

  dim3 grid(T_ / 128, B_);
  dim3 block(256);
  attn_score_wmma_kernel<<<grid, block, 0, stream>>>(X, g, WO, WG, v, out);
}
